// CustomBernsteinLayer_79499844649238
// MI455X (gfx1250) — matrix operand order repaired and checked by reading, not compiled, not run on hardware
//
#include <hip/hip_runtime.h>
#include <hip/hip_bf16.h>

// CDNA5 / gfx1250 WMMA kernel for the fused Bernstein-basis layer.
// out[b,o] = sum_{i,k} (x_a^k * x_b^(8-k))[b,i] * (coeffs[o,i,k]*weights[o,i])
// => GEMM: Ber[16384 x 2304] * WC^T[2304 x 256], K laid out k-major per
//    32-wide i-block (so each 32-K slab has a constant Bernstein exponent).

typedef __attribute__((ext_vector_type(16))) _Float16 v16h;
typedef __attribute__((ext_vector_type(8)))  float    v8f;

#define B_ROWS   16384
#define I_DIM    256
#define O_DIM    256
#define K_ORD    9            // order+1 (k = 0..8)
#define MBLK     128          // output rows per workgroup
#define AB_PITCH 33           // LDS pitch in float2 units (bank-stagger pad)
#define SLABS    72           // 8 i-blocks * 9 k values
#define FRAG_H   512          // halves per 32x16 B fragment
#define SLAB_H   8192         // halves per slab (16 fragments, 16 KB)

// x^k for constant k in [0,8]: folds to a minimal multiply chain after unroll.
__device__ __forceinline__ float ipowf(float x, int k) {
  float r = 1.0f, p = x;
  if (k & 1) r *= p;
  p *= p;
  if (k & 2) r *= p;
  p *= p;
  if (k & 4) r *= p;
  p *= p;
  if (k & 8) r *= p;
  return r;
}

// Branch-free tanh: 1 - 2/(e^{2x}+1), via v_exp_2f + v_rcp_f32.
__device__ __forceinline__ float fast_tanh(float v) {
  const float TWO_LOG2E = 2.885390081777927f;   // 2*log2(e)
  const float e = __builtin_amdgcn_exp2f(v * TWO_LOG2E);
  return 1.0f - 2.0f * __builtin_amdgcn_rcpf(e + 1.0f);
}

// Async DMA of one 16 KB WC slab into LDS: 256 threads x 4 x 16B, per-lane
// global_load_async_to_lds_b128 (tracked by ASYNCcnt, no VGPR round trip).
__device__ __forceinline__ void async_copy_slab(const _Float16* __restrict__ src,
                                                _Float16* dst, int tid) {
#pragma unroll
  for (int r = 0; r < 4; ++r) {
    const int e = tid + 256 * r;                               // 16B chunk id
    const uint32_t lofs = (uint32_t)(size_t)dst + (uint32_t)e * 16u;
    const uint64_t gadr = (uint64_t)(size_t)src + (uint64_t)e * 16u;
    asm volatile("global_load_async_to_lds_b128 %0, %1, off"
                 :: "v"(lofs), "v"(gadr)
                 : "memory");
  }
}

// ---------------------------------------------------------------------------
// Kernel 1: WC[o,i,k] = coeffs[o,i,k] * weights[o,i] -> f16, pre-swizzled into
// a 32x16 WMMA B-fragment lane layout in contiguous k order (same as A):
//   lane l: N = l&15, K_local = (l>>4)*16 + h   (h = half index 0..15)
// Fragment order: frag = (ib*9 + kk)*16 + ntile ; 512 halves each.
// ---------------------------------------------------------------------------
__global__ __launch_bounds__(256) void build_wc_kernel(
    const float* __restrict__ weights,   // [O, I]
    const float* __restrict__ coeffs,    // [O, I, 9]
    _Float16* __restrict__ wc)           // [72*16*512] halves (1.18 MB)
{
  const int tid  = threadIdx.x;
  const int l    = tid & 31;
  const int frag = blockIdx.x * 8 + (tid >> 5);   // 0..1151
  const int t    = frag & 15;                     // n-tile
  const int slab = frag >> 4;                     // 0..71
  const int ib   = slab / K_ORD;                  // i-block
  const int kk   = slab % K_ORD;                  // Bernstein index
  const int o    = t * 16 + (l & 15);
  const int ibase = ib * 32 + ((l >> 4) << 4);    // + h gives global i

  __align__(16) _Float16 tmp[16];
#pragma unroll
  for (int h = 0; h < 16; ++h) {
    const int i = ibase + h;
    const float v = coeffs[(size_t)(o * I_DIM + i) * K_ORD + kk] *
                    weights[(size_t)o * I_DIM + i];
    tmp[h] = (_Float16)v;
  }
  _Float16* dstp = wc + (size_t)frag * FRAG_H + l * 16;   // 32B per lane
  ((uint4*)dstp)[0] = ((const uint4*)tmp)[0];
  ((uint4*)dstp)[1] = ((const uint4*)tmp)[1];
}

// ---------------------------------------------------------------------------
// Kernel 2: WMMA GEMM. One WG = 256 threads (8 waves) -> 128 x 256 output.
// Wave grid 4(M) x 2(N): each wave owns 32 rows (2 A fragments) and 8 n-tiles.
// Three software pipelines overlap: (1) B slabs via double-buffered async DMA,
// (2) B fragments via bc/bn LDS-load rotation under the WMMA pair, (3) the
// next-k A fragments (pure VALU power chains) built *inside* the WMMA loop so
// they co-execute with the XDL pipe instead of serializing before it.
// ---------------------------------------------------------------------------
__global__ __launch_bounds__(256) void bern_gemm_kernel(
    const float* __restrict__ x,        // [B_ROWS, I_DIM]
    const _Float16* __restrict__ wc,    // pre-swizzled B fragments
    float* __restrict__ out)            // [B_ROWS, O_DIM]
{
  __shared__ __align__(16) float2   ab_lds[MBLK * AB_PITCH];  // (x_a, x_b)
  __shared__ __align__(32) _Float16 B_lds[2][SLAB_H];         // 2 x 16 KB

  const int tid    = threadIdx.x;
  const int l      = tid & 31;
  const int wv     = tid >> 5;
  const int mstrip = wv >> 1;               // 0..3 -> 32-row strip
  const int nhalf  = wv & 1;                // 0..1 -> 8-tile column group
  const int laneHi = l >> 4;
  const int row0   = mstrip * 32 + (l & 15);  // local row of A fragment 0
  const int Mbase  = blockIdx.x * MBLK;
  const _Float16* bbase = &B_lds[0][0] + (nhalf * 8 * 32 + l) * 16;
  const float2* ab0p = &ab_lds[row0 * AB_PITCH];
  const float2* ab1p = &ab_lds[(row0 + 16) * AB_PITCH];

  v8f acc0[8], acc1[8];
#pragma unroll
  for (int j = 0; j < 8; ++j) {
    acc0[j] = (v8f){0.f, 0.f, 0.f, 0.f, 0.f, 0.f, 0.f, 0.f};
    acc1[j] = (v8f){0.f, 0.f, 0.f, 0.f, 0.f, 0.f, 0.f, 0.f};
  }

  // Prologue: start DMA of slab 0 into buffer 0.
  async_copy_slab(wc, &B_lds[0][0], tid);
  int par = 0;

  for (int ib = 0; ib < 8; ++ib) {
    // Stage (x_a, x_b) for 128 rows x 32 i (safe: trailing barrier passed).
#pragma unroll
    for (int p = 0; p < 4; ++p) {
      const int rl = (tid >> 3) + p * 32;
      const int il = (tid & 7) * 4;
      const float4 xv = *(const float4*)&x[(size_t)(Mbase + rl) * I_DIM +
                                           ib * 32 + il];
      float t;
      t = fast_tanh(xv.x); ab_lds[rl * AB_PITCH + il + 0] = make_float2(t + 1.0f, 1.0f - t);
      t = fast_tanh(xv.y); ab_lds[rl * AB_PITCH + il + 1] = make_float2(t + 1.0f, 1.0f - t);
      t = fast_tanh(xv.z); ab_lds[rl * AB_PITCH + il + 2] = make_float2(t + 1.0f, 1.0f - t);
      t = fast_tanh(xv.w); ab_lds[rl * AB_PITCH + il + 3] = make_float2(t + 1.0f, 1.0f - t);
    }
    __syncthreads();   // (x_a,x_b) visible to all waves

    // Build the k=0 A fragments up front (b^8 chains, constant-folded).
    v16h a0c, a1c;
#pragma unroll
    for (int h = 0; h < 16; ++h) {
      const int Kl = h + (laneHi << 4);
      const float2 e0 = ab0p[Kl];
      const float2 e1 = ab1p[Kl];
      a0c[h] = (_Float16)(ipowf(e0.x, 0) * ipowf(e0.y, 8));
      a1c[h] = (_Float16)(ipowf(e1.x, 0) * ipowf(e1.y, 8));
    }

#pragma unroll
    for (int kk = 0; kk < K_ORD; ++kk) {
      const int s = ib * K_ORD + kk;
      // Kick off DMA of the *next* slab into the other buffer, then retire
      // only the current buffer's 4 ops (async ops complete in order) so the
      // new DMA stays in flight under the WMMAs.
      if (s + 1 < SLABS) {
        async_copy_slab(wc + (size_t)(s + 1) * SLAB_H, &B_lds[par ^ 1][0], tid);
        asm volatile("s_wait_asynccnt 0x4" ::: "memory");
      } else {
        asm volatile("s_wait_asynccnt 0x0" ::: "memory");
      }
      __syncthreads();   // publish current slab

      const _Float16* bp = bbase + (size_t)par * SLAB_H;
      v16h bc = *(const v16h*)(bp);        // fragment 0 in flight early

      // 16 WMMAs with two interleaved pipelines:
      //  - B fragment j+1 loads under the WMMA pair of j (bc/bn rotation)
      //  - A fragments for k+1 built 2 halves per step (VALU co-executes
      //    with the XDL WMMAs; ISA tracks 16-bit WMMA as TRANS class)
      v16h a0n, a1n;
#pragma unroll
      for (int j = 0; j < 8; ++j) {
        v16h bn;
        if (j < 7) bn = *(const v16h*)(bp + (size_t)(j + 1) * 32 * 16);
        acc0[j] = __builtin_amdgcn_wmma_f32_16x16x32_f16(
            false, a0c, false, bc, (short)0, acc0[j], false, false);
        acc1[j] = __builtin_amdgcn_wmma_f32_16x16x32_f16(
            false, a1c, false, bc, (short)0, acc1[j], false, false);
        if (kk + 1 < K_ORD) {
#pragma unroll
          for (int q = 0; q < 2; ++q) {
            const int h  = 2 * j + q;
            const int Kl = h + (laneHi << 4);
            const float2 e0 = ab0p[Kl];
            const float2 e1 = ab1p[Kl];
            a0n[h] = (_Float16)(ipowf(e0.x, kk + 1) * ipowf(e0.y, 7 - kk));
            a1n[h] = (_Float16)(ipowf(e1.x, kk + 1) * ipowf(e1.y, 7 - kk));
          }
        }
        if (j < 7) bc = bn;
      }

      __syncthreads();   // all waves done reading B_lds[par]
      par ^= 1;
      if (kk + 1 < K_ORD) { a0c = a0n; a1c = a1n; }
    }
  }

  // --- store: C/D layout: VGPR r -> M = 8*(l>>4) + r, N = l&15 --------------
  const int nl = l & 15;
  const int m0 = Mbase + mstrip * 32 + laneHi * 8;
#pragma unroll
  for (int j = 0; j < 8; ++j) {
    const int col = (nhalf * 8 + j) * 16 + nl;
#pragma unroll
    for (int r = 0; r < 8; ++r) {
      out[(size_t)(m0 + r) * O_DIM + col]      = acc0[j][r];
      out[(size_t)(m0 + 16 + r) * O_DIM + col] = acc1[j][r];
    }
  }
}

// ---------------------------------------------------------------------------
extern "C" void kernel_launch(void* const* d_in, const int* in_sizes, int n_in,
                              void* d_out, int out_size, void* d_ws,
                              size_t ws_size, hipStream_t stream) {
  const float* x       = (const float*)d_in[0];   // [16384, 256]
  const float* weights = (const float*)d_in[1];   // [256, 256]
  const float* coeffs  = (const float*)d_in[2];   // [256, 256, 9]
  float*       out     = (float*)d_out;           // [16384, 256]
  _Float16*    wc      = (_Float16*)d_ws;         // needs 72*16*512*2 = 1.18 MB

  // Phase 1: fold weights into coeffs, f16, pre-swizzled B fragments.
  build_wc_kernel<<<144, 256, 0, stream>>>(weights, coeffs, wc);
  // Phase 2: WMMA GEMM over the fused (i,k) axis.
  bern_gemm_kernel<<<B_ROWS / MBLK, 256, 0, stream>>>(x, wc, out);
}
